// EMSA_77524159693114
// MI455X (gfx1250) — compile-verified, matrix operand order checked by reading, not run on hardware
//
#include <hip/hip_runtime.h>
#include <hip/hip_bf16.h>

typedef __attribute__((ext_vector_type(16))) __bf16   bf16x16;
typedef __attribute__((ext_vector_type(8)))  float    f32x8;
typedef __attribute__((ext_vector_type(4)))  unsigned u32x4;
typedef __attribute__((ext_vector_type(8)))  int      i32x8;
typedef __attribute__((ext_vector_type(4)))  int      i32x4;

static constexpr int Bn  = 4;
static constexpr int NHn = 8;
static constexpr int NQn = 4096;   // 64*64
static constexpr int NKn = 1024;   // 32*32

union FragB16 {
    bf16x16 v;
    __bf16 e[16];
    unsigned u[8];
};

// K-pair base index for 16-bit A/B WMMA fragments (ISA 7.12.2):
// vgpr v in 0..7, each holds a packed bf16 pair.
__device__ __forceinline__ int kpb(int v, int lane) {
    return ((v & 4) ? 16 : 0) + ((lane & 16) ? 8 : 0) + 2 * (v & 3);
}

__device__ __forceinline__ f32x8 wmma_bf16(bf16x16 a, bf16x16 b, f32x8 c) {
    return __builtin_amdgcn_wmma_f32_16x16x32_bf16(false, a, false, b, (short)0, c, false, false);
}

// ---------------------------------------------------------------- converts
__global__ void cvt_f32_to_bf16(const float* __restrict__ in, __bf16* __restrict__ out, int n) {
    int i = blockIdx.x * blockDim.x + threadIdx.x;
    if (i < n) out[i] = (__bf16)in[i];
}

// in (K,N) f32 row-major -> out (N,K) bf16 (weight transpose for B-fragments)
__global__ __launch_bounds__(256) void cvt_transpose_to_bf16(
    const float* __restrict__ in, __bf16* __restrict__ out, int K, int N) {
    __shared__ float tile[32][33];
    int n0 = blockIdx.x * 32, k0 = blockIdx.y * 32;
    int tx = threadIdx.x & 31, ty = threadIdx.x >> 5;   // 32 x 8
    #pragma unroll
    for (int r = ty; r < 32; r += 8)
        tile[r][tx] = in[(size_t)(k0 + r) * N + n0 + tx];
    __syncthreads();
    #pragma unroll
    for (int r = ty; r < 32; r += 8)
        out[(size_t)(n0 + r) * K + k0 + tx] = (__bf16)tile[tx][r];
}

__global__ void zero_f32(float* __restrict__ p, int n) {
    int i = blockIdx.x * blockDim.x + threadIdx.x;
    if (i < n) p[i] = 0.0f;
}

// ------------------------------------------------- sr conv 3x3 stride2 pad1
__global__ __launch_bounds__(256) void sr_conv_kernel(
    const float* __restrict__ q, const float* __restrict__ w,
    const float* __restrict__ bias, float* __restrict__ out) {
    int b = blockIdx.y;
    int pix = blockIdx.x;              // 0..1023
    int yo = pix >> 5, xo = pix & 31;
    __shared__ float tin[9 * 256];
    for (int i = threadIdx.x; i < 9 * 256; i += 256) {
        int tap = i >> 8, ci = i & 255;
        int ky = tap / 3, kx = tap % 3;
        int iy = 2 * yo - 1 + ky, ix = 2 * xo - 1 + kx;
        float v = 0.0f;
        if (iy >= 0 && iy < 64 && ix >= 0 && ix < 64)
            v = q[((size_t)(b * NQn + iy * 64 + ix)) * 256 + ci];
        tin[tap * 256 + ci] = v;
    }
    __syncthreads();
    int co = threadIdx.x;
    float acc = bias[co];
    const float* wr = w + (size_t)co * 256 * 9;   // OIHW, (ci,ky,kx) contiguous
    for (int ci = 0; ci < 256; ++ci) {
        #pragma unroll
        for (int tap = 0; tap < 9; ++tap)
            acc = fmaf(tin[tap * 256 + ci], wr[ci * 9 + tap], acc);
    }
    out[((size_t)(b * NKn + pix)) * 256 + co] = acc;
}

// ------------------------------------------------------ layernorm over D=256
__global__ __launch_bounds__(256) void ln_kernel(
    const float* __restrict__ x, const float* __restrict__ g,
    const float* __restrict__ be, __bf16* __restrict__ out) {
    int b = blockIdx.y, p = blockIdx.x, c = threadIdx.x;
    size_t base = ((size_t)(b * NKn + p)) * 256;
    float v = x[base + c];
    float s = v, s2 = v * v;
    #pragma unroll
    for (int st = 16; st >= 1; st >>= 1) {
        s  += __shfl_xor(s, st, 32);
        s2 += __shfl_xor(s2, st, 32);
    }
    __shared__ float w1[8], w2[8];
    int wv = c >> 5, ln = c & 31;
    if (ln == 0) { w1[wv] = s; w2[wv] = s2; }
    __syncthreads();
    if (c == 0) {
        float t = 0, t2 = 0;
        for (int i = 0; i < 8; ++i) { t += w1[i]; t2 += w2[i]; }
        w1[0] = t; w2[0] = t2;
    }
    __syncthreads();
    float mean = w1[0] * (1.0f / 256.0f);
    float var  = w2[0] * (1.0f / 256.0f) - mean * mean;
    float y = (v - mean) * rsqrtf(var + 1e-5f) * g[c] + be[c];
    out[base + c] = (__bf16)y;
}

// --------------------------------------------- generic bf16 WMMA GEMM + bias
// A (batch, M, K) bf16 row-major, Bt (N, K) bf16 (pre-transposed), bias f32[N].
// One wave per 16x16 output tile. TRANSOUT stores C^T (N-major) for the V path.
template <bool F32OUT, bool TRANSOUT>
__global__ __launch_bounds__(32) void gemm_bf16_wmma(
    const __bf16* __restrict__ A, const __bf16* __restrict__ Bt,
    const float* __restrict__ bias, void* __restrict__ outP,
    int M, int N, int K, long long strideA, long long strideO) {
    int lane = threadIdx.x;
    int n0 = blockIdx.x * 16, m0 = blockIdx.y * 16;
    A += (long long)blockIdx.z * strideA;
    const __bf16* arow = &A[(size_t)(m0 + (lane & 15)) * K];
    const __bf16* brow = &Bt[(size_t)(n0 + (lane & 15)) * K];
    f32x8 acc = {0.f, 0.f, 0.f, 0.f, 0.f, 0.f, 0.f, 0.f};
    for (int k0 = 0; k0 < K; k0 += 32) {
        FragB16 a, bb;
        #pragma unroll
        for (int v = 0; v < 8; ++v) {
            int c = k0 + kpb(v, lane);
            a.u[v]  = *(const unsigned*)&arow[c];
            bb.u[v] = *(const unsigned*)&brow[c];
        }
        acc = wmma_bf16(a.v, bb.v, acc);
    }
    float bvv = bias[n0 + (lane & 15)];
    #pragma unroll
    for (int j = 0; j < 8; ++j) {
        int row = m0 + j + ((lane & 16) ? 8 : 0);
        int col = n0 + (lane & 15);
        float val = acc[j] + bvv;
        long long zo = (long long)blockIdx.z * strideO;
        if constexpr (F32OUT) {
            ((float*)outP)[zo + (size_t)row * N + col] = val;
        } else if constexpr (TRANSOUT) {
            ((__bf16*)outP)[zo + (size_t)col * M + row] = (__bf16)val;
        } else {
            ((__bf16*)outP)[zo + (size_t)row * N + col] = (__bf16)val;
        }
    }
}

// ----------------------------------------------------- fused mixed attention
// Block: (b, 16 queries); 8 waves, wave = output head o.
// Mixed score S_o = (q * tf_w[o, c/32] / sqrt(32)) @ Kcat^T (256-deep GEMM).
// 2-pass softmax; ssq accumulated for instance-norm variance; tf_b dropped
// (softmax-invariant). Vt is (b, NH*DV, NK) so P@V B-fragments are contiguous.
__global__ __launch_bounds__(256) void attn_kernel(
    const __bf16* __restrict__ Qbf, const __bf16* __restrict__ Kbf,
    const __bf16* __restrict__ Vt, const float* __restrict__ tfw,
    float* __restrict__ Oraw, float* __restrict__ ssqbuf) {
    int wave = threadIdx.x >> 5, lane = threadIdx.x & 31;
    int b = blockIdx.y, q0 = blockIdx.x * 16;
    int o = wave;
    __shared__ __align__(16) __bf16 qtile[16 * 256];
    __shared__ __align__(4)  __bf16 pstage[8 * 16 * 32];

#if __has_builtin(__builtin_amdgcn_tensor_load_to_lds) && __has_builtin(__builtin_amdgcn_s_wait_tensorcnt)
    // Tensor Data Mover: DMA the 16x256 bf16 Q tile into LDS (one descriptor).
    // amdgpu-toolchain (clang-23) 6-arg form: (g0, g1, g2, g3, g_extra, cpol).
    if (wave == 0) {
        unsigned lds_off = (unsigned)(size_t)&qtile[0];          // low 32b of LDS aperture addr
        unsigned long long ga =
            (unsigned long long)(const void*)(Qbf + ((size_t)(b * NQn + q0)) * 256);
        u32x4 g0 = { 1u,                                   // count=1 valid descriptor
                     lds_off,                              // lds_addr
                     (unsigned)ga,                         // global_addr[31:0]
                     ((unsigned)(ga >> 32) & 0x01FFFFFFu) | 0x80000000u }; // addr[56:32] | type=2
        i32x8 g1 = { 0x00010000,                           // data_size=1 (2 bytes)
                     (int)(256u << 16),                    // tensor_dim0[15:0]
                     (int)(16u << 16),                     // tensor_dim0 hi | tensor_dim1=16
                     (int)(256u << 16),                    // tensor_dim1 hi | tile_dim0=256
                     16,                                   // tile_dim1=16, tile_dim2=0
                     256,                                  // tensor_dim0_stride[31:0]
                     0, 0 };
        i32x4 gz  = { 0, 0, 0, 0 };
        i32x8 gz8 = { 0, 0, 0, 0, 0, 0, 0, 0 };
        __builtin_amdgcn_tensor_load_to_lds(g0, g1, gz, gz, gz8, 0);
        __builtin_amdgcn_s_wait_tensorcnt(0);
    }
    __syncthreads();
#else
    for (int i = threadIdx.x; i < 16 * 256; i += 256)
        qtile[i] = Qbf[((size_t)(b * NQn + q0 + (i >> 8))) * 256 + (i & 255)];
    __syncthreads();
#endif

    // Build 8 A-fragments (16x32 each) with per-head mixing scale folded in.
    const float isq = 0.17677669529663687f;   // 1/sqrt(32)
    FragB16 afr[8];
    #pragma unroll
    for (int f = 0; f < 8; ++f) {
        float sc = tfw[o * 8 + f] * isq;
        #pragma unroll
        for (int v = 0; v < 8; ++v) {
            int c = f * 32 + kpb(v, lane);
            int m = lane & 15;
            afr[f].e[2 * v]     = (__bf16)((float)qtile[m * 256 + c] * sc);
            afr[f].e[2 * v + 1] = (__bf16)((float)qtile[m * 256 + c + 1] * sc);
        }
    }

    const __bf16* krow = &Kbf[((size_t)(b * NKn)) * 256];

    // ---- pass 1: row max over all 1024 keys
    float rm[8];
    #pragma unroll
    for (int j = 0; j < 8; ++j) rm[j] = -3.0e38f;
    for (int kk = 0; kk < NKn; kk += 16) {
        if (kk + 16 < NKn)
            __builtin_prefetch(&krow[(size_t)(kk + 16 + (lane & 15)) * 256], 0, 1);
        f32x8 s = {0.f, 0.f, 0.f, 0.f, 0.f, 0.f, 0.f, 0.f};
        #pragma unroll
        for (int f = 0; f < 8; ++f) {
            FragB16 bk;
            #pragma unroll
            for (int v = 0; v < 8; ++v) {
                int c = f * 32 + kpb(v, lane);
                bk.u[v] = *(const unsigned*)&krow[(size_t)(kk + (lane & 15)) * 256 + c];
            }
            s = wmma_bf16(afr[f].v, bk.v, s);
        }
        #pragma unroll
        for (int j = 0; j < 8; ++j) rm[j] = fmaxf(rm[j], s[j]);
    }
    #pragma unroll
    for (int j = 0; j < 8; ++j)
        #pragma unroll
        for (int st = 1; st < 16; st <<= 1)
            rm[j] = fmaxf(rm[j], __shfl_xor(rm[j], st, 16));

    // ---- pass 2: exp, sums, and unnormalized P@V
    float ls[8], sq[8];
    #pragma unroll
    for (int j = 0; j < 8; ++j) { ls[j] = 0.f; sq[j] = 0.f; }
    f32x8 o0 = {0.f, 0.f, 0.f, 0.f, 0.f, 0.f, 0.f, 0.f};
    f32x8 o1 = {0.f, 0.f, 0.f, 0.f, 0.f, 0.f, 0.f, 0.f};
    for (int kk = 0; kk < NKn; kk += 32) {
        #pragma unroll
        for (int sub = 0; sub < 2; ++sub) {
            f32x8 s = {0.f, 0.f, 0.f, 0.f, 0.f, 0.f, 0.f, 0.f};
            #pragma unroll
            for (int f = 0; f < 8; ++f) {
                FragB16 bk;
                #pragma unroll
                for (int v = 0; v < 8; ++v) {
                    int c = f * 32 + kpb(v, lane);
                    bk.u[v] = *(const unsigned*)
                        &krow[(size_t)(kk + sub * 16 + (lane & 15)) * 256 + c];
                }
                s = wmma_bf16(afr[f].v, bk.v, s);
            }
            #pragma unroll
            for (int j = 0; j < 8; ++j) {
                float p = __expf(s[j] - rm[j]);
                ls[j] += p;
                sq[j] += p * p;
                int row = j + ((lane & 16) ? 8 : 0);
                pstage[o * 512 + row * 32 + sub * 16 + (lane & 15)] = (__bf16)p;
            }
        }
        __syncthreads();
        FragB16 pf, vf0, vf1;
        const __bf16* v0r = &Vt[((size_t)(b * 256 + o * 32 + (lane & 15))) * NKn + kk];
        const __bf16* v1r = &Vt[((size_t)(b * 256 + o * 32 + 16 + (lane & 15))) * NKn + kk];
        #pragma unroll
        for (int v = 0; v < 8; ++v) {
            int c = kpb(v, lane);
            pf.u[v]  = *(const unsigned*)&pstage[o * 512 + (lane & 15) * 32 + c];
            vf0.u[v] = *(const unsigned*)&v0r[c];
            vf1.u[v] = *(const unsigned*)&v1r[c];
        }
        o0 = wmma_bf16(pf.v, vf0.v, o0);
        o1 = wmma_bf16(pf.v, vf1.v, o1);
        __syncthreads();
    }
    #pragma unroll
    for (int j = 0; j < 8; ++j)
        #pragma unroll
        for (int st = 1; st < 16; st <<= 1) {
            ls[j] += __shfl_xor(ls[j], st, 16);
            sq[j] += __shfl_xor(sq[j], st, 16);
        }
    #pragma unroll
    for (int j = 0; j < 8; ++j) {
        int row = q0 + j + ((lane & 16) ? 8 : 0);
        float inv = 1.0f / ls[j];
        size_t base = ((size_t)(b * NQn + row)) * 256 + o * 32 + (lane & 15);
        Oraw[base]      = o0[j] * inv;
        Oraw[base + 16] = o1[j] * inv;
    }
    if ((lane & 15) == 0) {
        float loc = 0.f;
        #pragma unroll
        for (int j = 0; j < 8; ++j) loc += sq[j] / (ls[j] * ls[j]);
        atomicAdd(&ssqbuf[b * 8 + o], loc);
    }
}

// ----------------------------------- colsum of V over keys (Vt is N-major)
__global__ __launch_bounds__(256) void colsum_kernel(const __bf16* __restrict__ Vt,
                                                     float* __restrict__ cs) {
    int b = blockIdx.y, c = threadIdx.x;
    const __bf16* r = &Vt[((size_t)(b * 256 + c)) * NKn];
    float s = 0.f;
    for (int k = 0; k < NKn; ++k) s += (float)r[k];
    cs[b * 256 + c] = s;
}

// ---------------------- apply instance-norm scale/shift, emit bf16 for Wo GEMM
__global__ void norm_o_kernel(const float* __restrict__ Oraw,
                              const float* __restrict__ cs,
                              const float* __restrict__ ssq,
                              __bf16* __restrict__ Ohat) {
    int idx = blockIdx.x * blockDim.x + threadIdx.x;
    if (idx >= Bn * NQn * 256) return;
    int c = idx & 255;
    int b = idx >> 8; b /= NQn;
    int o = c >> 5;
    const float mu = 1.0f / (float)NKn;
    float var = ssq[b * 8 + o] / ((float)NQn * (float)NKn) - mu * mu;
    float s = rsqrtf(var + 1e-5f);
    Ohat[idx] = (__bf16)((Oraw[idx] - mu * cs[b * 256 + c]) * s);
}

// ---------------------------------------------------------------------------
extern "C" void kernel_launch(void* const* d_in, const int* in_sizes, int n_in,
                              void* d_out, int out_size, void* d_ws, size_t ws_size,
                              hipStream_t stream) {
    const float* query = (const float*)d_in[0];
    const float* Wq = (const float*)d_in[3];
    const float* bq = (const float*)d_in[4];
    const float* Wk = (const float*)d_in[5];
    const float* bk = (const float*)d_in[6];
    const float* Wv = (const float*)d_in[7];
    const float* bv = (const float*)d_in[8];
    const float* Wo = (const float*)d_in[9];
    const float* bo = (const float*)d_in[10];
    const float* sr_w = (const float*)d_in[11];
    const float* sr_b = (const float*)d_in[12];
    const float* ln_g = (const float*)d_in[13];
    const float* ln_b = (const float*)d_in[14];
    const float* tf_w = (const float*)d_in[15];

    char* w = (char*)d_ws;
    size_t off = 0;
    auto alloc = [&](size_t bytes) { size_t o = off; off += (bytes + 255) & ~(size_t)255; return o; };
    float*  xconv = (float*)(w + alloc((size_t)Bn * NKn * 256 * 4));
    __bf16* xln   = (__bf16*)(w + alloc((size_t)Bn * NKn * 256 * 2));
    __bf16* qin   = (__bf16*)(w + alloc((size_t)Bn * NQn * 256 * 2));   // reused as Ohat
    __bf16* Wqt   = (__bf16*)(w + alloc((size_t)256 * 256 * 2));
    __bf16* Wkt   = (__bf16*)(w + alloc((size_t)256 * 256 * 2));
    __bf16* Wvt   = (__bf16*)(w + alloc((size_t)256 * 256 * 2));
    __bf16* Wot   = (__bf16*)(w + alloc((size_t)256 * 256 * 2));
    __bf16* Qbf   = (__bf16*)(w + alloc((size_t)Bn * NQn * 256 * 2));
    __bf16* Kbf   = (__bf16*)(w + alloc((size_t)Bn * NKn * 256 * 2));
    __bf16* Vt    = (__bf16*)(w + alloc((size_t)Bn * NKn * 256 * 2));   // (b, NH*DV, NK)
    float*  Oraw  = (float*)(w + alloc((size_t)Bn * NQn * 256 * 4));
    float*  csum  = (float*)(w + alloc((size_t)Bn * 256 * 4));
    float*  ssq   = (float*)(w + alloc((size_t)Bn * NHn * 4));
    __bf16* Ohat  = qin;   // alias: qin is dead after the Q projection

    int nq = Bn * NQn * 256;
    cvt_f32_to_bf16<<<(nq + 255) / 256, 256, 0, stream>>>(query, qin, nq);
    cvt_transpose_to_bf16<<<dim3(8, 8), 256, 0, stream>>>(Wq, Wqt, 256, 256);
    cvt_transpose_to_bf16<<<dim3(8, 8), 256, 0, stream>>>(Wk, Wkt, 256, 256);
    cvt_transpose_to_bf16<<<dim3(8, 8), 256, 0, stream>>>(Wv, Wvt, 256, 256);
    cvt_transpose_to_bf16<<<dim3(8, 8), 256, 0, stream>>>(Wo, Wot, 256, 256);

    sr_conv_kernel<<<dim3(NKn, Bn), 256, 0, stream>>>(query, sr_w, sr_b, xconv);
    ln_kernel<<<dim3(NKn, Bn), 256, 0, stream>>>(xconv, ln_g, ln_b, xln);

    // Q = qin @ Wq + bq  (bf16 out)
    gemm_bf16_wmma<false, false><<<dim3(16, NQn / 16, Bn), 32, 0, stream>>>(
        qin, Wqt, bq, Qbf, NQn, 256, 256,
        (long long)NQn * 256, (long long)NQn * 256);
    // K = xln @ Wk + bk (bf16 out) ; V = xln @ Wv + bv (bf16, transposed out)
    gemm_bf16_wmma<false, false><<<dim3(16, NKn / 16, Bn), 32, 0, stream>>>(
        xln, Wkt, bk, Kbf, NKn, 256, 256,
        (long long)NKn * 256, (long long)NKn * 256);
    gemm_bf16_wmma<false, true><<<dim3(16, NKn / 16, Bn), 32, 0, stream>>>(
        xln, Wvt, bv, Vt, NKn, 256, 256,
        (long long)NKn * 256, (long long)NKn * 256);

    zero_f32<<<1, 64, 0, stream>>>(ssq, Bn * NHn);
    attn_kernel<<<dim3(NQn / 16, Bn), 256, 0, stream>>>(Qbf, Kbf, Vt, tf_w, Oraw, ssq);

    colsum_kernel<<<dim3(1, Bn), 256, 0, stream>>>(Vt, csum);
    norm_o_kernel<<<(Bn * NQn * 256 + 255) / 256, 256, 0, stream>>>(Oraw, csum, ssq, Ohat);

    // out = Ohat @ Wo + bo  (f32 out -> d_out)
    gemm_bf16_wmma<true, false><<<dim3(16, NQn / 16, Bn), 32, 0, stream>>>(
        Ohat, Wot, bo, (float*)d_out, NQn, 256, 256,
        (long long)NQn * 256, (long long)NQn * 256);
}
